// PPO_GNN_LSTM_54700703482083
// MI455X (gfx1250) — matrix operand order repaired and checked by reading, not compiled, not run on hardware
//
#include <hip/hip_runtime.h>
#include <hip/hip_bf16.h>
#include <math.h>

#define S_DIM   8
#define N_NODES 5000
#define F_IN    32
#define E_EDGES 80000
#define EP_EDGES (E_EDGES + N_NODES)   // + self loops
#define EMB     128
#define HEADS   4
#define CH      32
#define G4      (4 * EMB)              // 512 LSTM gate cols
#define NEG_SLOPE 0.2f

typedef __attribute__((ext_vector_type(16))) __bf16 v16bf;
typedef __attribute__((ext_vector_type(8)))  float  v8f;

union V16 { v16bf v; float4 q[2]; };

__device__ __forceinline__ unsigned short f2bf(float f) {
  unsigned int u = __float_as_uint(f);
  unsigned int r = (u + 0x7FFFu + ((u >> 16) & 1u)) >> 16;   // round-nearest-even
  return (unsigned short)r;
}

__device__ __forceinline__ float sigf(float x) { return 1.0f / (1.0f + __expf(-x)); }

__device__ __forceinline__ void atomicMaxF(float* addr, float val) {
  if (val >= 0.0f) atomicMax((int*)addr, __float_as_int(val));
  else             atomicMin((unsigned int*)addr, __float_as_uint(val));
}

// ---------------------------------------------------------------- converts
__global__ void k_f32_to_bf16(const float* __restrict__ src,
                              unsigned short* __restrict__ dst, int n) {
  int i = blockIdx.x * blockDim.x + threadIdx.x;
  if (i < n) dst[i] = f2bf(src[i]);
}

__global__ void k_addvec(const float* __restrict__ a, const float* __restrict__ b,
                         float* __restrict__ o, int n) {
  int i = blockIdx.x * blockDim.x + threadIdx.x;
  if (i < n) o[i] = a[i] + b[i];
}

// ---------------------------------------------------------------- WMMA GEMM
// Y[rows][ncols] = Ab[rows][DIN](bf16) @ Wb[ncols][DIN]^T(bf16) + bias | +Y(accum)
// One wave computes a 16x32 output slab (two 16x16 tiles) so each A fragment
// feeds two v_wmma ops. bf16 A/B, f32 accumulate. 32-bit addressing.
template <int DIN>
__global__ void k_gemm(const unsigned short* __restrict__ Ab,
                       const unsigned short* __restrict__ Wb,
                       const float* __restrict__ bias,
                       float* __restrict__ Y,
                       int rows, int ncols, int accum) {
  int gid  = blockIdx.x * blockDim.x + threadIdx.x;
  int w    = gid >> 5;
  int lane = threadIdx.x & 31;
  int MT = (rows + 15) >> 4;
  int NG = ncols >> 5;                     // 32-wide column groups
  if (w >= MT * NG) return;                // uniform per-wave: EXEC stays all-ones
  int mt = w / NG, ng = w % NG;
  int hi = lane >> 4, lm = lane & 15;
  int row = mt * 16 + lm;
  int rl  = row < rows ? row : rows - 1;   // clamp: garbage rows never stored
  int col0 = ng * 32 + lm;
  int col1 = col0 + 16;

  int  m0   = mt * 16 + 8 * hi;            // first of this lane-half's 8 D rows
  bool full = (mt * 16 + 16) <= rows;      // all but the last M tile
  float* yp = Y + m0 * ncols + col0;

  v8f c0, c1;
  if (accum) {
    if (full) {
#pragma unroll
      for (int r = 0; r < 8; ++r) { c0[r] = yp[r * ncols]; c1[r] = yp[r * ncols + 16]; }
    } else {
#pragma unroll
      for (int r = 0; r < 8; ++r) {
        int orow = m0 + r; if (orow >= rows) orow = rows - 1;
        const float* p = Y + orow * ncols;
        c0[r] = p[col0]; c1[r] = p[col1];
      }
    }
  } else {
    float b0 = bias[col0], b1 = bias[col1];
#pragma unroll
    for (int r = 0; r < 8; ++r) { c0[r] = b0; c1[r] = b1; }
  }

  const unsigned short* arow  = Ab + rl * DIN;
  const unsigned short* wrow0 = Wb + col0 * DIN;
  const unsigned short* wrow1 = Wb + col1 * DIN;
#pragma unroll
  for (int k0 = 0; k0 < DIN; k0 += 32) {
    V16 a, b0v, b1v;
    // A bf16 16x32 layout: lanes hold M=lm; K chunks [k0+8*hi, +7], [k0+16+8*hi, +7]
    a.q[0]   = *(const float4*)(arow + k0 + 8 * hi);
    a.q[1]   = *(const float4*)(arow + k0 + 16 + 8 * hi);
    // B bf16 32x16 layout: lanes hold N=lm; K chunks [k0+8*hi, +7], [k0+16+8*hi, +7]
    b0v.q[0] = *(const float4*)(wrow0 + k0 + 8 * hi);
    b0v.q[1] = *(const float4*)(wrow0 + k0 + 16 + 8 * hi);
    b1v.q[0] = *(const float4*)(wrow1 + k0 + 8 * hi);
    b1v.q[1] = *(const float4*)(wrow1 + k0 + 16 + 8 * hi);
    c0 = __builtin_amdgcn_wmma_f32_16x16x32_bf16(false, a.v, false, b0v.v,
                                                 (short)0, c0, false, false);
    c1 = __builtin_amdgcn_wmma_f32_16x16x32_bf16(false, a.v, false, b1v.v,
                                                 (short)0, c1, false, false);
  }

  if (full) {
#pragma unroll
    for (int r = 0; r < 8; ++r) { yp[r * ncols] = c0[r]; yp[r * ncols + 16] = c1[r]; }
  } else {
#pragma unroll
    for (int r = 0; r < 8; ++r) {
      int orow = m0 + r;
      if (orow < rows) { yp[r * ncols] = c0[r]; yp[r * ncols + 16] = c1[r]; }
    }
  }
}

// ---------------------------------------------------------------- GAT edge ops
__global__ void k_init_seg(float* __restrict__ mseg, float* __restrict__ dseg,
                           float* __restrict__ agg) {
  int i = blockIdx.x * blockDim.x + threadIdx.x;
  if (i < N_NODES * HEADS) { mseg[i] = -__builtin_inff(); dseg[i] = 0.0f; }
  if (i < N_NODES * EMB) agg[i] = 0.0f;
}

__global__ void k_edge_logit(const int* __restrict__ ei,
                             const float* __restrict__ xl, const float* __restrict__ xr,
                             const float* __restrict__ att,
                             float* __restrict__ logit, float* __restrict__ mseg) {
  int t = blockIdx.x * blockDim.x + threadIdx.x;
  if (t >= EP_EDGES * HEADS) return;
  int e = t >> 2, h = t & 3;
  int src, dst;
  if (e < E_EDGES) { src = ei[e]; dst = ei[E_EDGES + e]; }
  else             { src = e - E_EDGES; dst = src; }
  const float* pl = xl + src * EMB + h * CH;
  const float* pr = xr + dst * EMB + h * CH;
  const float* pa = att + h * CH;
  float acc = 0.0f;
#pragma unroll 8
  for (int cc = 0; cc < CH; ++cc) {
    float v = pl[cc] + pr[cc];
    v = v > 0.0f ? v : NEG_SLOPE * v;
    acc += v * pa[cc];
  }
  logit[t] = acc;
  atomicMaxF(mseg + dst * HEADS + h, acc);
}

__global__ void k_edge_exp(const int* __restrict__ ei,
                           const float* __restrict__ mseg,
                           float* __restrict__ logit, float* __restrict__ dseg) {
  int t = blockIdx.x * blockDim.x + threadIdx.x;
  if (t >= EP_EDGES * HEADS) return;
  int e = t >> 2, h = t & 3;
  int dst = (e < E_EDGES) ? ei[E_EDGES + e] : (e - E_EDGES);
  float ex = __expf(logit[t] - mseg[dst * HEADS + h]);
  logit[t] = ex;
  atomicAdd(dseg + dst * HEADS + h, ex);
}

__global__ void k_alpha(const int* __restrict__ ei, const float* __restrict__ dseg,
                        float* __restrict__ exv) {
  int t = blockIdx.x * blockDim.x + threadIdx.x;
  if (t >= EP_EDGES * HEADS) return;
  int e = t >> 2, h = t & 3;
  int dst = (e < E_EDGES) ? ei[E_EDGES + e] : (e - E_EDGES);
  exv[t] = exv[t] / dseg[dst * HEADS + h];   // normalize once per (edge, head)
}

__global__ void k_edge_scatter(const int* __restrict__ ei,
                               const float* __restrict__ xl,
                               const float* __restrict__ alpha,
                               float* __restrict__ agg) {
  int t = blockIdx.x * blockDim.x + threadIdx.x;
  if (t >= EP_EDGES * EMB) return;
  int e = t >> 7, d = t & 127;
  int h = d >> 5;
  int src, dst;
  if (e < E_EDGES) { src = ei[e]; dst = ei[E_EDGES + e]; }
  else             { src = e - E_EDGES; dst = src; }
  atomicAdd(agg + dst * EMB + d, xl[src * EMB + d] * alpha[e * HEADS + h]);
}

__global__ void k_finish_layer(const float* __restrict__ agg, const float* __restrict__ bias,
                               unsigned short* __restrict__ xb_next,
                               float* __restrict__ mu_f, unsigned short* __restrict__ mu_b,
                               int last) {
  int i = blockIdx.x * blockDim.x + threadIdx.x;
  if (i >= N_NODES * EMB) return;
  int d = i & (EMB - 1);
  float v = agg[i] + bias[d];
  if (!last) {
    v = v > 0.0f ? v : 0.0f;                // inter-layer ReLU
    xb_next[i] = f2bf(v);
  } else {
    mu_f[i] = v;                            // no ReLU after last GAT layer
    mu_b[i] = f2bf(v);
  }
}

// ---------------------------------------------------------------- LSTM
__global__ void k_lstm_init(const float* __restrict__ h0, const float* __restrict__ c0,
                            unsigned short* __restrict__ hb, float* __restrict__ cst) {
  int i = blockIdx.x * blockDim.x + threadIdx.x;
  if (i >= N_NODES * EMB) return;
  hb[i] = f2bf(h0[i]);
  cst[i] = c0[i];
}

__global__ void k_lstm_elem(const float* __restrict__ gates,
                            float* __restrict__ cst, unsigned short* __restrict__ hb,
                            float* __restrict__ mu_f, unsigned short* __restrict__ mu_b) {
  int i = blockIdx.x * blockDim.x + threadIdx.x;
  if (i >= N_NODES * EMB) return;
  int n = i >> 7, d = i & 127;
  const float* g = gates + n * G4;
  float gi = g[d], gf = g[EMB + d], gg = g[2 * EMB + d], go = g[3 * EMB + d];
  float cc = sigf(gf) * cst[i] + sigf(gi) * tanhf(gg);
  float hh = sigf(go) * tanhf(cc);
  cst[i] = cc;
  hb[i] = f2bf(hh);       // recurrent input (bf16 for WMMA)
  mu_f[i] = hh;           // scan output replaces mu[t]
  mu_b[i] = f2bf(hh);
}

// ---------------------------------------------------------------- head
__global__ void k_meanpool(const float* __restrict__ mu, float* __restrict__ mu_mean) {
  int s = blockIdx.x, d = threadIdx.x;     // 128 threads
  const float* p = mu + (size_t)s * N_NODES * EMB + d;
  float acc = 0.0f;
  for (int n = 0; n < N_NODES; ++n) acc += p[n * EMB];
  mu_mean[s * EMB + d] = acc * (1.0f / (float)N_NODES);
}

__global__ void k_gs(const float* __restrict__ mu_mean, const float* __restrict__ W6,
                     const float* __restrict__ b6, const float* __restrict__ w5,
                     float* __restrict__ gterm) {
  __shared__ float sm[EMB];
  int s = blockIdx.x, d = threadIdx.x;
  const float* mm = mu_mean + s * EMB;
  const float* wr = W6 + d * EMB;
  float acc = b6[d];
  for (int k = 0; k < EMB; ++k) acc += mm[k] * wr[k];
  acc = acc > 0.0f ? acc : 0.0f;
  sm[d] = acc * w5[d];
  __syncthreads();
  for (int st = 64; st > 0; st >>= 1) { if (d < st) sm[d] += sm[d + st]; __syncthreads(); }
  if (d == 0) gterm[s] = sm[0];
}

__global__ void k_logits(const float* __restrict__ la, const float* __restrict__ w5,
                         const float* __restrict__ gterm, const float* __restrict__ b5,
                         const unsigned char* __restrict__ reach, float* __restrict__ logits) {
  int i = blockIdx.x * blockDim.x + threadIdx.x;
  if (i >= S_DIM * N_NODES) return;
  int s = i / N_NODES;
  const float* p = la + i * EMB;
  float acc = 0.0f;
#pragma unroll 8
  for (int d = 0; d < EMB; ++d) {
    float v = p[d]; v = v > 0.0f ? v : 0.0f;
    acc += v * w5[EMB + d];
  }
  float lg = acc + gterm[s] + b5[0];
  logits[i] = reach[i] ? lg : -__builtin_inff();
}

__global__ void k_softmax(const float* __restrict__ logits, float* __restrict__ out) {
  __shared__ float sm[1024];
  int s = blockIdx.x, t = threadIdx.x;
  const float* p = logits + s * N_NODES;
  float mx = -__builtin_inff();
  for (int n = t; n < N_NODES; n += 1024) mx = fmaxf(mx, p[n]);
  sm[t] = mx; __syncthreads();
  for (int st = 512; st > 0; st >>= 1) { if (t < st) sm[t] = fmaxf(sm[t], sm[t + st]); __syncthreads(); }
  float gmx = sm[0]; __syncthreads();
  float sum = 0.0f;
  for (int n = t; n < N_NODES; n += 1024) sum += __expf(p[n] - gmx);
  sm[t] = sum; __syncthreads();
  for (int st = 512; st > 0; st >>= 1) { if (t < st) sm[t] += sm[t + st]; __syncthreads(); }
  float inv = 1.0f / sm[0];
  for (int n = t; n < N_NODES; n += 1024) out[s * N_NODES + n] = __expf(p[n] - gmx) * inv;
}

// ---------------------------------------------------------------- host
extern "C" void kernel_launch(void* const* d_in, const int* in_sizes, int n_in,
                              void* d_out, int out_size, void* d_ws, size_t ws_size,
                              hipStream_t stream) {
  (void)in_sizes; (void)n_in; (void)out_size; (void)ws_size;
  auto cdiv = [](long a, long b) { return (int)((a + b - 1) / b); };
  // waves needed for a GEMM of (rows x ncols): one wave per 16x32 output slab
  auto gemm_grid = [&](long rows, long ncols) {
    return cdiv(((rows + 15) / 16) * (ncols / 32) * 32 /*threads per wave*/, 256);
  };

  const float*          nfm   = (const float*)d_in[0];
  const int*            ei    = (const int*)d_in[1];
  const unsigned char*  reach = (const unsigned char*)d_in[2];
  const float*          h0    = (const float*)d_in[3];
  const float*          c0    = (const float*)d_in[4];
  int ix = 5;
  const float *Wl[5], *bl[5], *Wr[5], *br[5], *att[5], *gbias[5];
  for (int l = 0; l < 5; ++l) {
    Wl[l]    = (const float*)d_in[ix++];
    bl[l]    = (const float*)d_in[ix++];
    Wr[l]    = (const float*)d_in[ix++];
    br[l]    = (const float*)d_in[ix++];
    att[l]   = (const float*)d_in[ix++];
    gbias[l] = (const float*)d_in[ix++];
  }
  const float* W_ih = (const float*)d_in[ix++];
  const float* b_ih = (const float*)d_in[ix++];
  const float* W_hh = (const float*)d_in[ix++];
  const float* b_hh = (const float*)d_in[ix++];
  const float* W6   = (const float*)d_in[ix++];
  const float* b6   = (const float*)d_in[ix++];
  const float* W7   = (const float*)d_in[ix++];
  const float* b7   = (const float*)d_in[ix++];
  const float* w5   = (const float*)d_in[ix++];
  const float* b5   = (const float*)d_in[ix++];
  float* out = (float*)d_out;

  // ---- workspace carve-out (~76.5 MB; fits in 192MB L2)
  size_t off = 0;
  auto alloc = [&](size_t bytes) -> void* {
    void* p = (char*)d_ws + off;
    off += (bytes + 255) & ~(size_t)255;
    return p;
  };
  unsigned short *wbl[5], *wbr[5];
  for (int l = 0; l < 5; ++l) {
    int din = (l == 0) ? F_IN : EMB;
    wbl[l] = (unsigned short*)alloc((size_t)EMB * din * 2);
    wbr[l] = (unsigned short*)alloc((size_t)EMB * din * 2);
  }
  unsigned short* wb_ih = (unsigned short*)alloc((size_t)G4 * EMB * 2);
  unsigned short* wb_hh = (unsigned short*)alloc((size_t)G4 * EMB * 2);
  unsigned short* wb7   = (unsigned short*)alloc((size_t)EMB * EMB * 2);
  float* bsum   = (float*)alloc(G4 * 4);
  unsigned short* xb = (unsigned short*)alloc((size_t)N_NODES * EMB * 2);
  float* xl   = (float*)alloc((size_t)N_NODES * EMB * 4);
  float* xr   = (float*)alloc((size_t)N_NODES * EMB * 4);
  float* mseg = (float*)alloc((size_t)N_NODES * HEADS * 4);
  float* dseg = (float*)alloc((size_t)N_NODES * HEADS * 4);
  float* exv  = (float*)alloc((size_t)EP_EDGES * HEADS * 4);
  float* agg  = (float*)alloc((size_t)N_NODES * EMB * 4);
  float* mu_f = (float*)alloc((size_t)S_DIM * N_NODES * EMB * 4);
  unsigned short* mu_b = (unsigned short*)alloc((size_t)S_DIM * N_NODES * EMB * 2);
  float* cst  = (float*)alloc((size_t)N_NODES * EMB * 4);
  unsigned short* hb = (unsigned short*)alloc((size_t)N_NODES * EMB * 2);
  float* gates = (float*)alloc((size_t)N_NODES * G4 * 4);
  float* la    = (float*)alloc((size_t)S_DIM * N_NODES * EMB * 4);
  float* mu_mean = (float*)alloc((size_t)S_DIM * EMB * 4);
  float* gterm   = (float*)alloc(S_DIM * 4);
  float* logitsb = (float*)alloc((size_t)S_DIM * N_NODES * 4);

  // ---- weight conversion to bf16 (once per launch)
  for (int l = 0; l < 5; ++l) {
    int n = EMB * ((l == 0) ? F_IN : EMB);
    k_f32_to_bf16<<<cdiv(n, 256), 256, 0, stream>>>(Wl[l], wbl[l], n);
    k_f32_to_bf16<<<cdiv(n, 256), 256, 0, stream>>>(Wr[l], wbr[l], n);
  }
  k_f32_to_bf16<<<cdiv(G4 * EMB, 256), 256, 0, stream>>>(W_ih, wb_ih, G4 * EMB);
  k_f32_to_bf16<<<cdiv(G4 * EMB, 256), 256, 0, stream>>>(W_hh, wb_hh, G4 * EMB);
  k_f32_to_bf16<<<cdiv(EMB * EMB, 256), 256, 0, stream>>>(W7, wb7, EMB * EMB);
  k_addvec<<<cdiv(G4, 256), 256, 0, stream>>>(b_ih, b_hh, bsum, G4);

  const int gat_grid = gemm_grid(N_NODES, EMB);
  const int ne_grid  = cdiv((long)EP_EDGES * HEADS, 256);
  const int sc_grid  = cdiv((long)EP_EDGES * EMB, 256);
  const int nd_grid  = cdiv((long)N_NODES * EMB, 256);

  // ---- GAT: 5 layers per sample
  for (int s = 0; s < S_DIM; ++s) {
    k_f32_to_bf16<<<cdiv((long)N_NODES * F_IN, 256), 256, 0, stream>>>(
        nfm + (size_t)s * N_NODES * F_IN, xb, N_NODES * F_IN);
    for (int l = 0; l < 5; ++l) {
      int last = (l == 4);
      if (l == 0) {
        k_gemm<F_IN><<<gat_grid, 256, 0, stream>>>(xb, wbl[l], bl[l], xl, N_NODES, EMB, 0);
        k_gemm<F_IN><<<gat_grid, 256, 0, stream>>>(xb, wbr[l], br[l], xr, N_NODES, EMB, 0);
      } else {
        k_gemm<EMB><<<gat_grid, 256, 0, stream>>>(xb, wbl[l], bl[l], xl, N_NODES, EMB, 0);
        k_gemm<EMB><<<gat_grid, 256, 0, stream>>>(xb, wbr[l], br[l], xr, N_NODES, EMB, 0);
      }
      k_init_seg<<<nd_grid, 256, 0, stream>>>(mseg, dseg, agg);
      k_edge_logit<<<ne_grid, 256, 0, stream>>>(ei, xl, xr, att[l], exv, mseg);
      k_edge_exp<<<ne_grid, 256, 0, stream>>>(ei, mseg, exv, dseg);
      k_alpha<<<ne_grid, 256, 0, stream>>>(ei, dseg, exv);
      k_edge_scatter<<<sc_grid, 256, 0, stream>>>(ei, xl, exv, agg);
      k_finish_layer<<<nd_grid, 256, 0, stream>>>(
          agg, gbias[l], xb,
          mu_f + (size_t)s * N_NODES * EMB, mu_b + (size_t)s * N_NODES * EMB, last);
    }
  }

  // ---- LSTM over S steps
  k_lstm_init<<<nd_grid, 256, 0, stream>>>(h0, c0, hb, cst);
  const int lstm_grid = gemm_grid(N_NODES, G4);
  for (int t = 0; t < S_DIM; ++t) {
    k_gemm<EMB><<<lstm_grid, 256, 0, stream>>>(
        mu_b + (size_t)t * N_NODES * EMB, wb_ih, bsum, gates, N_NODES, G4, 0);
    k_gemm<EMB><<<lstm_grid, 256, 0, stream>>>(hb, wb_hh, bsum, gates, N_NODES, G4, 1);
    k_lstm_elem<<<nd_grid, 256, 0, stream>>>(
        gates, cst, hb, mu_f + (size_t)t * N_NODES * EMB, mu_b + (size_t)t * N_NODES * EMB);
  }

  // ---- head
  k_meanpool<<<S_DIM, EMB, 0, stream>>>(mu_f, mu_mean);
  k_gs<<<S_DIM, EMB, 0, stream>>>(mu_mean, W6, b6, w5, gterm);
  k_gemm<EMB><<<gemm_grid((long)S_DIM * N_NODES, EMB), 256, 0, stream>>>(
      mu_b, wb7, b7, la, S_DIM * N_NODES, EMB, 0);
  k_logits<<<cdiv((long)S_DIM * N_NODES, 256), 256, 0, stream>>>(la, w5, gterm, b5, reach, logitsb);
  k_softmax<<<S_DIM, 1024, 0, stream>>>(logitsb, out);
}
